// ConvolutionLayer_28767690948702
// MI455X (gfx1250) — compile-verified, matrix operand order checked by reading, not run on hardware
//
#include <hip/hip_runtime.h>

// Problem constants (match reference)
#define B_   2
#define L_   4096
#define DIN  256
#define DOUT 256
#define K_   16

typedef __attribute__((ext_vector_type(16))) _Float16 v16h;
typedef __attribute__((ext_vector_type(8)))  _Float16 v8h;
typedef __attribute__((ext_vector_type(8)))  float    v8f;
typedef __attribute__((ext_vector_type(4)))  float    v4f;
typedef __attribute__((ext_vector_type(4)))  unsigned int v4u;
typedef __attribute__((ext_vector_type(8)))  int      v8i;
typedef __attribute__((ext_vector_type(4)))  int      v4i;

__device__ __forceinline__ float2 cmulf(float2 a, float2 b) {
    return make_float2(a.x * b.x - a.y * b.y, a.x * b.y + a.y * b.x);
}

// ---------------------------------------------------------------------------
// Tensor Data Mover: 1-D row DMA (4096 x f32, 16KB) global -> LDS.
// D# built per CDNA5 ISA 8.3/8.4: group0 = {count=1, lds_addr, global_addr,
// type=2}; group1 = {data_size=4B, tensor_dim0=tile_dim0=4096, tile_dim1=1,
// dim0_stride=4096}. Groups 2/3 zero (<=2D tensor). Tracked via TENSORcnt.
// Toolchain here is the 6-arg clang-23 builtin form.
// ---------------------------------------------------------------------------
__device__ __forceinline__ void tdm_load_row16k(const void* gptr, unsigned lds_addr) {
    const unsigned long long ga = (unsigned long long)(uintptr_t)gptr;
    v4u g0;
    g0[0] = 1u;                                               // count=1 valid D#
    g0[1] = lds_addr;                                         // LDS byte address
    g0[2] = (unsigned)(ga & 0xffffffffu);                     // global_addr lo
    g0[3] = (unsigned)((ga >> 32) & 0x01ffffffu) | (2u << 30); // ga hi | type=2
    v8i g1;
    g1[0] = (int)(2u << 16);        // workgroup_mask=0, data_size=2 (4 bytes)
    g1[1] = (int)(4096u << 16);     // tensor_dim0[15:0] in bits 63:48
    g1[2] = (int)(1u << 16);        // tensor_dim0 hi=0, tensor_dim1=1
    g1[3] = (int)(4096u << 16);     // tensor_dim1 hi=0, tile_dim0=4096
    g1[4] = 1;                      // tile_dim1=1, tile_dim2=0
    g1[5] = 4096;                   // tensor_dim0_stride lo
    g1[6] = 0;                      // stride hi, tensor_dim1_stride lo
    g1[7] = 0;
    const v4i z4 = {0, 0, 0, 0};
    const v8i z8 = {0, 0, 0, 0, 0, 0, 0, 0};
    __builtin_amdgcn_tensor_load_to_lds(g0, g1, z4, z4, z8, 0);
}

// ---------------------------------------------------------------------------
// In-LDS 4096-point radix-2 DIT FFT, 256 threads, dir=+1 fwd / -1 inv.
// Twiddles computed on the fly with __sincosf (keeps LDS small).
// ---------------------------------------------------------------------------
__device__ __forceinline__ void fft4096(float2* s, float dir) {
    const int tid = threadIdx.x;
    __syncthreads();
    // bit-reversal permutation (12 bits)
    for (int i = tid; i < 4096; i += 256) {
        int j = __brev(i) >> 20;
        if (i < j) { float2 t = s[i]; s[i] = s[j]; s[j] = t; }
    }
    __syncthreads();
    for (int len = 2; len <= 4096; len <<= 1) {
        const int   half = len >> 1;
        const float fac  = dir * (-6.283185307179586f) / (float)len;
        for (int m = tid; m < 2048; m += 256) {
            const int pos = m & (half - 1);
            const int i1  = ((m & ~(half - 1)) << 1) | pos;
            const int i2  = i1 + half;
            float sn, cs;
            __sincosf(fac * (float)pos, &sn, &cs);
            const float2 w = make_float2(cs, sn);
            const float2 a = s[i1];
            const float2 b = s[i2];
            const float2 t = cmulf(w, b);
            s[i1] = make_float2(a.x + t.x, a.y + t.y);
            s[i2] = make_float2(a.x - t.x, a.y - t.y);
        }
        __syncthreads();
    }
}

// ---------------------------------------------------------------------------
// Precision prep: f32 -> f16 copies (u as-is; M transposed to (k, p, d) so
// both WMMA fragments load as contiguous b128)
// ---------------------------------------------------------------------------
__global__ void __launch_bounds__(256) cvt_u_kernel(const float* __restrict__ u,
                                                    _Float16* __restrict__ uh, int n) {
    int i = blockIdx.x * blockDim.x + threadIdx.x;
    if (i < n) uh[i] = (_Float16)u[i];
}

__global__ void __launch_bounds__(256) cvt_mt_kernel(const float* __restrict__ M,
                                                     _Float16* __restrict__ Mt) {
    int i = blockIdx.x * blockDim.x + threadIdx.x;  // over K*DIN*DOUT
    int p = i & 255;
    int d = (i >> 8) & 255;
    int k = i >> 16;
    Mt[((size_t)(k * DOUT + p) << 8) + d] = (_Float16)M[i];
}

// ---------------------------------------------------------------------------
// FFT of the 16 eigenvector filters -> evF[k][f] (complex, full 4096 bins)
// ---------------------------------------------------------------------------
__global__ void __launch_bounds__(256) evfft_kernel(const float* __restrict__ ev,
                                                    float2* __restrict__ evF) {
    __shared__ float2 s[4096];
    const int k = blockIdx.x;
    for (int i = threadIdx.x; i < 4096; i += 256)
        s[i] = make_float2(ev[k * 4096 + i], 0.0f);
    fft4096(s, 1.0f);
    for (int i = threadIdx.x; i < 4096; i += 256)
        evF[k * 4096 + i] = s[i];
}

// ---------------------------------------------------------------------------
// WMMA fragment helpers (16-bit A 16x32 layout: lanes 0-15 carry K 0-7/16-23,
// lanes 16-31 carry K 8-15/24-31 -> two contiguous b128 loads per fragment)
// ---------------------------------------------------------------------------
__device__ __forceinline__ v16h ldfrag(const _Float16* p) {
    const v8h lo = *(const v8h*)(p);
    const v8h hi = *(const v8h*)(p + 16);
    v16h r;
    #pragma unroll
    for (int i = 0; i < 8; ++i) { r[i] = lo[i]; r[i + 8] = hi[i]; }
    return r;
}

__device__ __forceinline__ void store_tile(float* dst, v8f c, float lamk) {
    v4f lo = { c[0] * lamk, c[1] * lamk, c[2] * lamk, c[3] * lamk };
    v4f hi = { c[4] * lamk, c[5] * lamk, c[6] * lamk, c[7] * lamk };
    *(v4f*)(dst)     = lo;
    *(v4f*)(dst + 4) = hi;
}

// ---------------------------------------------------------------------------
// WMMA GEMM: v[b,k,p,l] = lambda_k * sum_d u[b,l,d] * M[k,d,p]
// 256 thr = 8 waves. Each wave: one 16-row tile x FOUR 16-col tiles, sharing
// the A fragment across 4 wmma per K-step -> 32 v_wmma / wave, 2.5 loads/wmma.
// Grid (256 row-pairs, 1, 16 k).
// ---------------------------------------------------------------------------
__global__ void __launch_bounds__(256) gemm_v_kernel(const _Float16* __restrict__ Uh,
                                                     const _Float16* __restrict__ Mt,
                                                     const float* __restrict__ lam,
                                                     float* __restrict__ v) {
    const int lane    = threadIdx.x & 31;
    const int wave    = threadIdx.x >> 5;
    const int rowTile = blockIdx.x * 2 + (wave >> 2);   // 0..511 over B*L/16
    const int colBase = (wave & 3) * 64;                // 4 tiles of 16 cols
    const int k       = blockIdx.z;

    const int row  = rowTile * 16 + (lane & 15);
    const int ksel = (lane >> 4) * 8;

    const _Float16* Arow = Uh + (size_t)row * DIN;
    const _Float16* Bc   = Mt + ((size_t)(k * DOUT + colBase + (lane & 15))) * DIN;

    v8f c0 = {}, c1 = {}, c2 = {}, c3 = {};
    #pragma unroll
    for (int kk = 0; kk < DIN; kk += 32) {
        const int  kb = kk + ksel;
        const v16h a  = ldfrag(Arow + kb);
        v16h b;
        b  = ldfrag(Bc + kb);
        c0 = __builtin_amdgcn_wmma_f32_16x16x32_f16(false, a, false, b, (short)0, c0, false, false);
        b  = ldfrag(Bc + (size_t)16 * DIN + kb);
        c1 = __builtin_amdgcn_wmma_f32_16x16x32_f16(false, a, false, b, (short)0, c1, false, false);
        b  = ldfrag(Bc + (size_t)32 * DIN + kb);
        c2 = __builtin_amdgcn_wmma_f32_16x16x32_f16(false, a, false, b, (short)0, c2, false, false);
        b  = ldfrag(Bc + (size_t)48 * DIN + kb);
        c3 = __builtin_amdgcn_wmma_f32_16x16x32_f16(false, a, false, b, (short)0, c3, false, false);
    }

    // C layout: VGPR j holds row m = j + 8*(lane>=16), column n = lane&15.
    // Rows are consecutive l -> two b128 stores per tile into (b,k,p, L).
    const float lamk = lam[k];
    const int   bIdx = rowTile >> 8;
    const int   lrow = (rowTile & 255) * 16 + (lane >> 4) * 8;
    const size_t planeBase = ((size_t)(bIdx * K_ + k) * DOUT) << 12;
    const int    colLane   = colBase + (lane & 15);
    store_tile(v + planeBase + (((size_t)colLane +  0) << 12) + lrow, c0, lamk);
    store_tile(v + planeBase + (((size_t)colLane + 16) << 12) + lrow, c1, lamk);
    store_tile(v + planeBase + (((size_t)colLane + 32) << 12) + lrow, c2, lamk);
    store_tile(v + planeBase + (((size_t)colLane + 48) << 12) + lrow, c3, lamk);
}

// ---------------------------------------------------------------------------
// Frequency-domain convolution + k-reduction + inverse FFT.
// One block per (b, p). The NEXT k's 16KB v row is DMA'd into an LDS staging
// buffer by the Tensor Data Mover while the current FFT runs; evF spectrum of
// k+1 is prefetched into L2/L0 (one cacheline per lane).
// ---------------------------------------------------------------------------
__global__ void __launch_bounds__(256) conv_kernel(const float* __restrict__ v,
                                                   const float2* __restrict__ evF,
                                                   float* __restrict__ out) {
    __shared__ float2 s[4096];       // 32 KB complex working set
    __shared__ float  rowbuf[4096];  // 16 KB TDM staging buffer
    const int b = blockIdx.x >> 8;
    const int p = blockIdx.x & 255;
    const unsigned rowbufLds = (unsigned)(uintptr_t)rowbuf;

    // Kick off DMA of row k=0 (one wave issues; TDM ignores EXEC).
    if (threadIdx.x < 32)
        tdm_load_row16k(v + (((size_t)(b * K_ + 0) * DOUT + p) << 12), rowbufLds);

    float2 acc[16];
    #pragma unroll
    for (int t = 0; t < 16; ++t) acc[t] = make_float2(0.0f, 0.0f);

    for (int k = 0; k < K_; ++k) {
        __builtin_amdgcn_s_wait_tensorcnt(0);   // row k landed in rowbuf
        __syncthreads();
        for (int i = threadIdx.x; i < 4096; i += 256)
            s[i] = make_float2(rowbuf[i], 0.0f);
        __syncthreads();                        // all done reading rowbuf
        if (k + 1 < K_) {
            if (threadIdx.x < 32)               // overlap DMA of k+1 with FFT
                tdm_load_row16k(v + (((size_t)(b * K_ + k + 1) * DOUT + p) << 12),
                                rowbufLds);
            // warm L2/L0 for next spectrum: one 128B line per lane
            __builtin_prefetch(evF + (size_t)(k + 1) * 4096 + threadIdx.x * 16, 0, 3);
        }

        fft4096(s, 1.0f);
        #pragma unroll
        for (int t = 0; t < 16; ++t) {
            const int i = threadIdx.x + t * 256;
            const float2 d = s[i];
            const float2 e = evF[k * 4096 + i];
            const float2 m = cmulf(d, e);
            acc[t].x += m.x;
            acc[t].y += m.y;
        }
        __syncthreads();   // done reading s before next k overwrites it
    }

    #pragma unroll
    for (int t = 0; t < 16; ++t) s[threadIdx.x + t * 256] = acc[t];
    fft4096(s, -1.0f);

    const float inv = 1.0f / 4096.0f;
    for (int i = threadIdx.x; i < 4096; i += 256)
        out[((size_t)b * L_ + i) * DOUT + p] = s[i].x * inv;
}

// ---------------------------------------------------------------------------
extern "C" void kernel_launch(void* const* d_in, const int* in_sizes, int n_in,
                              void* d_out, int out_size, void* d_ws, size_t ws_size,
                              hipStream_t stream) {
    const float* u   = (const float*)d_in[0];   // (B, L, DIN)
    const float* ev  = (const float*)d_in[1];   // (K, L)
    const float* lam = (const float*)d_in[2];   // (K,)
    const float* M   = (const float*)d_in[3];   // (K, DIN, DOUT)
    float* out = (float*)d_out;                 // (B, L, DOUT)

    char* ws = (char*)d_ws;
    const size_t vBytes  = (size_t)B_ * K_ * DOUT * L_ * sizeof(float);    // 134 MB
    const size_t uhBytes = (size_t)B_ * L_ * DIN * sizeof(_Float16);       // 4 MB
    const size_t mtBytes = (size_t)K_ * DIN * DOUT * sizeof(_Float16);     // 2 MB
    float*    v   = (float*)(ws);
    _Float16* Uh  = (_Float16*)(ws + vBytes);
    _Float16* Mt  = (_Float16*)(ws + vBytes + uhBytes);
    float2*   evF = (float2*)(ws + vBytes + uhBytes + mtBytes);            // 512 KB

    const int nU = B_ * L_ * DIN;
    cvt_u_kernel <<<nU / 256, 256, 0, stream>>>(u, Uh, nU);
    cvt_mt_kernel<<<(K_ * DIN * DOUT) / 256, 256, 0, stream>>>(M, Mt);
    evfft_kernel <<<K_, 256, 0, stream>>>(ev, evF);
    gemm_v_kernel<<<dim3(256, 1, K_), 256, 0, stream>>>(Uh, Mt, lam, v);
    conv_kernel  <<<B_ * DOUT, 256, 0, stream>>>(v, evF, out);
}
